// SteeringAttention_22471268892911
// MI455X (gfx1250) — compile-verified, matrix operand order checked by reading, not run on hardware
//
#include <hip/hip_runtime.h>
#include <math.h>

// ---------------- problem constants ----------------
#define S_LEN   1024
#define NH      32
#define NKV     8
#define HD      128
#define HID     4096
#define KVW     1024          // NKV*HD
#define GROUPS  4             // NH/NKV
#define SCALE_QK 0.088388347648318447f   // 1/sqrt(128)

// ---------------- CDNA5 WMMA types ----------------
typedef __attribute__((ext_vector_type(16))) __bf16 v16bf;
typedef __attribute__((ext_vector_type(8)))  __bf16 bf16x8;
typedef __attribute__((ext_vector_type(8)))  float  v8f;
typedef __attribute__((ext_vector_type(4)))  unsigned int v4u;
typedef __attribute__((ext_vector_type(8)))  int v8i;
typedef __attribute__((ext_vector_type(4)))  int v4i;

// Tensor Data Mover availability (this toolchain: 6-arg builtin)
#ifndef USE_TDM
#if defined(__has_builtin)
#if __has_builtin(__builtin_amdgcn_tensor_load_to_lds)
#define USE_TDM 1
#else
#define USE_TDM 0
#endif
#else
#define USE_TDM 0
#endif
#endif

__device__ __forceinline__ __bf16 f2bf(float f) {
  unsigned u = __builtin_bit_cast(unsigned, f);
  unsigned r = (u + 0x7fffu + ((u >> 16) & 1u)) >> 16;
  return __builtin_bit_cast(__bf16, (unsigned short)r);
}
__device__ __forceinline__ float bf2f(__bf16 b) {
  unsigned u = ((unsigned)__builtin_bit_cast(unsigned short, b)) << 16;
  return __builtin_bit_cast(float, u);
}
__device__ __forceinline__ v8f zero8() {
  v8f z;
#pragma unroll
  for (int i = 0; i < 8; ++i) z[i] = 0.0f;
  return z;
}

// Load a 16x32 bf16 fragment in the CDNA5 16-bit A/B VGPR layout:
//   lanes 0-15 : row = lane,    K = {0..7, 16..23}
//   lanes 16-31: row = lane-16, K = {8..15, 24..31}
__device__ __forceinline__ v16bf load_frag(const __bf16* base, int stride) {
  const int lane = threadIdx.x & 31;
  const int half = lane >> 4;
  const int l    = lane & 15;
  const __bf16* p = base + l * stride + (half ? 8 : 0);
  bf16x8 lo = *(const bf16x8*)(p);
  bf16x8 hi = *(const bf16x8*)(p + 16);
  v16bf r;
#pragma unroll
  for (int i = 0; i < 8; ++i) { r[i] = lo[i]; r[8 + i] = hi[i]; }
  return r;
}

#if USE_TDM
// 2D tensor-DMA of bf16 tiles into LDS. Dims/strides in elements (data_size=2B).
// D# bit packing per CDNA5 ISA 8.3/8.4.
__device__ __forceinline__ void tdm_load_2d(unsigned lds_addr, const void* gptr,
                                            unsigned tile_d0, unsigned tile_d1,
                                            unsigned tensor_d0, unsigned tensor_d1,
                                            unsigned long long stride0) {
  unsigned long long ga = (unsigned long long)gptr;
  v4u g0;
  g0[0] = 1u;                                   // count=1, user mode, no gather
  g0[1] = lds_addr;                             // LDS byte address
  g0[2] = (unsigned)(ga & 0xffffffffu);         // global_addr[31:0]
  g0[3] = (unsigned)((ga >> 32) & 0x01ffffffu)  // global_addr[56:32]
        | (2u << 30);                           // type = 2 ("image")
  v8i g1;
  g1[0] = (int)(1u << 16);                      // wg_mask=0, data_size=1 (2B)
  g1[1] = (int)((tensor_d0 & 0xffffu) << 16);   // tensor_dim0[15:0] at bit 48
  g1[2] = (int)((tensor_d0 >> 16) | ((tensor_d1 & 0xffffu) << 16));
  g1[3] = (int)((tensor_d1 >> 16) | ((tile_d0 & 0xffffu) << 16));
  g1[4] = (int)(tile_d1 & 0xffffu);             // tile_dim1; tile_dim2=0
  g1[5] = (int)(stride0 & 0xffffffffu);         // tensor_dim0_stride[31:0]
  g1[6] = (int)((stride0 >> 32) & 0xffffu);     // stride[47:32]; dim1_stride=0
  g1[7] = 0;
  v4i gz4 = {0, 0, 0, 0};
  v8i gz8 = {0, 0, 0, 0, 0, 0, 0, 0};
  __builtin_amdgcn_tensor_load_to_lds(g0, g1, gz4, gz4, gz8, 0);
}
__device__ __forceinline__ unsigned lds_of(const void* p) {
  return (unsigned)(size_t)p;
}
#endif

// ---------------- fp32 -> bf16 convert ----------------
__global__ void cvt_f32_bf16(const float* __restrict__ in,
                             unsigned short* __restrict__ out, int n) {
  int i = blockIdx.x * blockDim.x + threadIdx.x;
  if (i < n) out[i] = __builtin_bit_cast(unsigned short, f2bf(in[i]));
}

// ---------------- GEMM: C[M,N] = A[M,K] @ W[N,K]^T  (bf16 in, bf16/f32 out)
// one wave per block; 32 rows x 64 cols; ping-pong pipelined K loop (step 64):
// two fragment stages alternate roles so no register copies are needed and
// each s_wait_loadcnt lands after a burst of 8 v_wmma issues.
__global__ void gemm_bf16_xwt(const unsigned short* __restrict__ A_,
                              const unsigned short* __restrict__ W_,
                              void* __restrict__ Cout,
                              int M, int N, int K, int out_f32) {
  const __bf16* A = (const __bf16*)A_;
  const __bf16* W = (const __bf16*)W_;
  const int n0 = blockIdx.x * 64;
  const int m0 = blockIdx.y * 32;
  const int lane = threadIdx.x & 31;
  const int half = lane >> 4;
  const int l    = lane & 15;

  const __bf16* Ap0 = A + (size_t)m0 * K;
  const __bf16* Ap1 = A + (size_t)(m0 + 16) * K;

  v8f acc[2][4];
#pragma unroll
  for (int r = 0; r < 2; ++r)
#pragma unroll
    for (int t = 0; t < 4; ++t) acc[r][t] = zero8();

  // stage A fragments at k = 0
  v16bf a0A = load_frag(Ap0, K);
  v16bf a1A = load_frag(Ap1, K);
  v16bf bA[4];
#pragma unroll
  for (int t = 0; t < 4; ++t) bA[t] = load_frag(W + (size_t)(n0 + t * 16) * K, K);

  for (int k = 0; k < K; k += 64) {
    // stage B loads at k+32 (issue before consuming stage A)
    v16bf a0B = load_frag(Ap0 + k + 32, K);
    v16bf a1B = load_frag(Ap1 + k + 32, K);
    v16bf bB[4];
#pragma unroll
    for (int t = 0; t < 4; ++t)
      bB[t] = load_frag(W + (size_t)(n0 + t * 16) * K + k + 32, K);
    if (k + 128 < K)  // pull upcoming A rows toward the WGP
      __builtin_prefetch((const void*)(Ap0 + (size_t)l * K + k + 128), 0, 3);

    // WMMA burst on stage A
#pragma unroll
    for (int t = 0; t < 4; ++t) {
      acc[0][t] = __builtin_amdgcn_wmma_f32_16x16x32_bf16(
          false, a0A, false, bA[t], (short)0, acc[0][t], false, false);
      acc[1][t] = __builtin_amdgcn_wmma_f32_16x16x32_bf16(
          false, a1A, false, bA[t], (short)0, acc[1][t], false, false);
    }

    // stage A loads at k+64 (clamped dummy reload on the last iteration)
    const int kn = (k + 64 < K) ? (k + 64) : 0;
    a0A = load_frag(Ap0 + kn, K);
    a1A = load_frag(Ap1 + kn, K);
#pragma unroll
    for (int t = 0; t < 4; ++t)
      bA[t] = load_frag(W + (size_t)(n0 + t * 16) * K + kn, K);

    // WMMA burst on stage B
#pragma unroll
    for (int t = 0; t < 4; ++t) {
      acc[0][t] = __builtin_amdgcn_wmma_f32_16x16x32_bf16(
          false, a0B, false, bB[t], (short)0, acc[0][t], false, false);
      acc[1][t] = __builtin_amdgcn_wmma_f32_16x16x32_bf16(
          false, a1B, false, bB[t], (short)0, acc[1][t], false, false);
    }
  }

  if (out_f32) {
    float* C = (float*)Cout;
#pragma unroll
    for (int r = 0; r < 2; ++r)
#pragma unroll
      for (int t = 0; t < 4; ++t)
#pragma unroll
        for (int i = 0; i < 8; ++i) {
          int row = m0 + r * 16 + i + half * 8;
          int col = n0 + t * 16 + l;
          C[(size_t)row * N + col] = acc[r][t][i];
        }
  } else {
    unsigned short* C = (unsigned short*)Cout;
#pragma unroll
    for (int r = 0; r < 2; ++r)
#pragma unroll
      for (int t = 0; t < 4; ++t)
#pragma unroll
        for (int i = 0; i < 8; ++i) {
          int row = m0 + r * 16 + i + half * 8;
          int col = n0 + t * 16 + l;
          C[(size_t)row * N + col] =
              __builtin_bit_cast(unsigned short, f2bf(acc[r][t][i]));
        }
  }
}

// ---------------- RoPE (+ optional gamma scale for K), in place, bf16 ------
__global__ void rope_scale(unsigned short* __restrict__ X_, int nh,
                           const float* __restrict__ prior, int do_scale) {
  __bf16* X = (__bf16*)X_;
  int idx = blockIdx.x * blockDim.x + threadIdx.x;
  int total = S_LEN * nh * (HD / 2);
  if (idx >= total) return;
  int i  = idx & 63;               // rotary pair index 0..63
  int hh = (idx >> 6) % nh;
  int s  = idx / (64 * nh);
  float invf = __expf(-(float)i * (logf(10000.0f) / 64.0f));
  float sinv, cosv;
  __sincosf((float)s * invf, &sinv, &cosv);
  __bf16* row = X + (size_t)s * nh * HD + hh * HD;
  float x1 = bf2f(row[i]);
  float x2 = bf2f(row[i + 64]);
  float g = 1.0f;
  if (do_scale) g = fminf(fmaxf(1.0f + 0.5f * prior[s], 0.5f), 2.0f);
  row[i]      = f2bf((x1 * cosv - x2 * sinv) * g);
  row[i + 64] = f2bf((x2 * cosv + x1 * sinv) * g);
}

// ---------------- V: eta scale + transpose into VT[KVW][S] ----------------
__global__ void v_scale_transpose(const unsigned short* __restrict__ V_,
                                  unsigned short* __restrict__ VT_,
                                  const float* __restrict__ prior) {
  int idx = blockIdx.x * blockDim.x + threadIdx.x;
  if (idx >= S_LEN * KVW) return;
  int s = idx / KVW;
  int c = idx - s * KVW;
  float eta = fminf(fmaxf(1.0f + 0.5f * prior[s], 0.5f), 2.0f);
  float v = bf2f(__builtin_bit_cast(__bf16, V_[idx])) * eta;
  VT_[c * S_LEN + s] = __builtin_bit_cast(unsigned short, f2bf(v));
}

// ---------------- fused flash attention, one wave = (head, 16 q rows) ------
// Score: S = (Q K^T)/sqrt(D) + m1*clip(prior_k) + causal mask
// Online softmax with column weights w_k = 1 + m2*0.5*prior_k (m2 in {0,1}),
// out = (sum e*w*v) / (sum e*w)  — exact fold of the two reference stages.
// K and V^T chunk tiles are double-buffered in LDS and DMA'd by the TDM:
// chunk i+1's DMAs are issued before chunk i is consumed (tensor ops from one
// wave complete in order, so s_wait_tensorcnt(2) retires the older pair).
__global__ void attn_fused(const unsigned short* __restrict__ Q_,
                           const unsigned short* __restrict__ K_,
                           const unsigned short* __restrict__ VT_,
                           const float* __restrict__ prior,
                           const float* __restrict__ hm1,
                           const float* __restrict__ hm2,
                           unsigned short* __restrict__ O_) {
  const __bf16* Q  = (const __bf16*)Q_;
  const __bf16* Kb = (const __bf16*)K_;
  const __bf16* VT = (const __bf16*)VT_;
  __shared__ __bf16 P[16 * 32];             // 16x32 softmax tile, row-major
#if USE_TDM
  __shared__ __bf16 Ktile[2][32 * 128];     // keys x head-dims, double buffered
  __shared__ __bf16 Vtile[2][128 * 32];     // V^T: head-dims x keys
#endif

  const int h   = blockIdx.y;
  const int q0  = blockIdx.x * 16;
  const int hk  = h / GROUPS;
  const int lane = threadIdx.x & 31;
  const int half = lane >> 4;
  const int l    = lane & 15;
  const float m1h = hm1[h];
  const float m2h = hm2[h];

  // Q fragments (16 rows x 128 head dims = 4 K-chunks of 32)
  v16bf qf[4];
#pragma unroll
  for (int dc = 0; dc < 4; ++dc)
    qf[dc] = load_frag(Q + (size_t)q0 * HID + h * HD + dc * 32, HID);

  v8f oacc[8];
#pragma unroll
  for (int d = 0; d < 8; ++d) oacc[d] = zero8();
  float Mr[8], Lr[8];
#pragma unroll
  for (int i = 0; i < 8; ++i) { Mr[i] = -3.0e38f; Lr[i] = 0.0f; }

  const int nchunks = (q0 + 15) / 32 + 1;   // causal: chunks of 32 keys

#if USE_TDM
  // prologue: DMA chunk 0 into buffer 0
  tdm_load_2d(lds_of(Ktile[0]), Kb + hk * HD,
              /*tile*/ 128, 32, /*tensor*/ 128, 32, /*stride*/ KVW);
  tdm_load_2d(lds_of(Vtile[0]), VT + (size_t)(hk * HD) * S_LEN,
              /*tile*/ 32, 128, /*tensor*/ 32, 128, /*stride*/ S_LEN);
#endif

  for (int c = 0; c < nchunks; ++c) {
    const int k0  = c * 32;
    const int cur = c & 1;
#if USE_TDM
    const int nxt = cur ^ 1;
    if (c + 1 < nchunks) {
      // WAR guard: buffer `nxt` was last read in chunk c-1
      asm volatile("s_wait_dscnt 0x0" ::: "memory");
      tdm_load_2d(lds_of(Ktile[nxt]), Kb + (size_t)(k0 + 32) * KVW + hk * HD,
                  128, 32, 128, 32, KVW);
      tdm_load_2d(lds_of(Vtile[nxt]), VT + (size_t)(hk * HD) * S_LEN + k0 + 32,
                  32, 128, 32, 128, S_LEN);
      __builtin_amdgcn_s_wait_tensorcnt(2);   // older pair (chunk c) complete
    } else {
      __builtin_amdgcn_s_wait_tensorcnt(0);
    }
#endif

    // ---- scores for two 16x16 key tiles ----
    v8f sacc[2];
#pragma unroll
    for (int t = 0; t < 2; ++t) {
      sacc[t] = zero8();
#pragma unroll
      for (int dc = 0; dc < 4; ++dc) {
#if USE_TDM
        v16bf b = load_frag(&Ktile[cur][(t * 16) * 128 + dc * 32], 128);
#else
        v16bf b = load_frag(Kb + (size_t)(k0 + t * 16) * KVW + hk * HD + dc * 32, KVW);
#endif
        sacc[t] = __builtin_amdgcn_wmma_f32_16x16x32_bf16(
            false, qf[dc], false, b, (short)0, sacc[t], false, false);
      }
    }

    // per-lane column constants for this chunk
    const int c0 = k0 + l, c1 = k0 + 16 + l;
    const float p0 = prior[c0], p1 = prior[c1];
    const float b0 = m1h * fminf(fmaxf(p0, -5.0f), 5.0f);
    const float b1 = m1h * fminf(fmaxf(p1, -5.0f), 5.0f);
    const float w0 = 1.0f + m2h * 0.5f * p0;
    const float w1 = 1.0f + m2h * 0.5f * p1;

    float f[8];
#pragma unroll
    for (int i = 0; i < 8; ++i) {
      const int row = q0 + i + half * 8;
      float s0 = sacc[0][i] * SCALE_QK + b0;
      float s1 = sacc[1][i] * SCALE_QK + b1;
      s0 = (c0 > row) ? (s0 - 1.0e9f) : s0;   // causal additive mask
      s1 = (c1 > row) ? (s1 - 1.0e9f) : s1;
      // row max across the 16 lanes of this half
      float mx = fmaxf(s0, s1);
#pragma unroll
      for (int off = 1; off < 16; off <<= 1)
        mx = fmaxf(mx, __shfl_xor(mx, off, 32));
      const float Mn = fmaxf(Mr[i], mx);
      const float fr = __expf(Mr[i] - Mn);
      Mr[i] = Mn;
      const float e0 = __expf(s0 - Mn) * w0;
      const float e1 = __expf(s1 - Mn) * w1;
      float rs = e0 + e1;
#pragma unroll
      for (int off = 1; off < 16; off <<= 1)
        rs += __shfl_xor(rs, off, 32);
      Lr[i] = Lr[i] * fr + rs;
      f[i] = fr;
      P[(i + half * 8) * 32 + l]      = f2bf(e0);
      P[(i + half * 8) * 32 + 16 + l] = f2bf(e1);
    }

    // rescale running output accumulators
#pragma unroll
    for (int d = 0; d < 8; ++d)
#pragma unroll
      for (int i = 0; i < 8; ++i) oacc[d][i] *= f[i];

    __syncthreads();
    v16bf pf = load_frag(P, 32);          // P tile back in A-layout via LDS
#pragma unroll
    for (int d = 0; d < 8; ++d) {
#if USE_TDM
      v16bf b = load_frag(&Vtile[cur][(d * 16) * 32], 32);
#else
      v16bf b = load_frag(VT + (size_t)(hk * HD + d * 16) * S_LEN + k0, S_LEN);
#endif
      oacc[d] = __builtin_amdgcn_wmma_f32_16x16x32_bf16(
          false, pf, false, b, (short)0, oacc[d], false, false);
    }
    __syncthreads();
  }

  // normalize + store bf16 [S, HID] with head h at cols h*HD..
#pragma unroll
  for (int d = 0; d < 8; ++d)
#pragma unroll
    for (int i = 0; i < 8; ++i) {
      int row = q0 + i + half * 8;
      int col = h * HD + d * 16 + l;
      O_[(size_t)row * HID + col] =
          __builtin_bit_cast(unsigned short, f2bf(oacc[d][i] / Lr[i]));
    }
}

// ---------------- host side ----------------
extern "C" void kernel_launch(void* const* d_in, const int* in_sizes, int n_in,
                              void* d_out, int out_size, void* d_ws, size_t ws_size,
                              hipStream_t stream) {
  (void)in_sizes; (void)n_in; (void)out_size; (void)ws_size;
  const float* hidden = (const float*)d_in[0];
  // d_in[1] attention_mask, d_in[2] position_ids: causal mask / arange done in-kernel
  const float* Wq   = (const float*)d_in[3];
  const float* Wk   = (const float*)d_in[4];
  const float* Wv   = (const float*)d_in[5];
  const float* Wo   = (const float*)d_in[6];
  const float* prior = (const float*)d_in[7];
  const float* hm1  = (const float*)d_in[8];
  const float* hm2  = (const float*)d_in[9];

  size_t off = 0;
  auto alloc = [&](size_t bytes) {
    void* p = (char*)d_ws + off;
    off += (bytes + 255) & ~(size_t)255;
    return p;
  };
  unsigned short* hid_bf = (unsigned short*)alloc((size_t)S_LEN * HID * 2);
  unsigned short* wq_bf  = (unsigned short*)alloc((size_t)HID * HID * 2);
  unsigned short* wk_bf  = (unsigned short*)alloc((size_t)KVW * HID * 2);
  unsigned short* wv_bf  = (unsigned short*)alloc((size_t)KVW * HID * 2);
  unsigned short* wo_bf  = (unsigned short*)alloc((size_t)HID * HID * 2);
  unsigned short* Qb     = (unsigned short*)alloc((size_t)S_LEN * HID * 2);
  unsigned short* Kbuf   = (unsigned short*)alloc((size_t)S_LEN * KVW * 2);
  unsigned short* Vbuf   = (unsigned short*)alloc((size_t)S_LEN * KVW * 2);
  unsigned short* VTb    = (unsigned short*)alloc((size_t)S_LEN * KVW * 2);
  unsigned short* Ob     = (unsigned short*)alloc((size_t)S_LEN * HID * 2);

  // 1) fp32 -> bf16
  cvt_f32_bf16<<<(S_LEN * HID) / 256, 256, 0, stream>>>(hidden, hid_bf, S_LEN * HID);
  cvt_f32_bf16<<<(HID * HID) / 256, 256, 0, stream>>>(Wq, wq_bf, HID * HID);
  cvt_f32_bf16<<<(KVW * HID) / 256, 256, 0, stream>>>(Wk, wk_bf, KVW * HID);
  cvt_f32_bf16<<<(KVW * HID) / 256, 256, 0, stream>>>(Wv, wv_bf, KVW * HID);
  cvt_f32_bf16<<<(HID * HID) / 256, 256, 0, stream>>>(Wo, wo_bf, HID * HID);

  // 2) projections (WMMA GEMMs, 32x64 tile per wave)
  gemm_bf16_xwt<<<dim3(HID / 64, S_LEN / 32), 32, 0, stream>>>(
      hid_bf, wq_bf, Qb, S_LEN, HID, HID, 0);
  gemm_bf16_xwt<<<dim3(KVW / 64, S_LEN / 32), 32, 0, stream>>>(
      hid_bf, wk_bf, Kbuf, S_LEN, KVW, HID, 0);
  gemm_bf16_xwt<<<dim3(KVW / 64, S_LEN / 32), 32, 0, stream>>>(
      hid_bf, wv_bf, Vbuf, S_LEN, KVW, HID, 0);

  // 3) RoPE (Q plain, K with gamma), V eta-scale + transpose
  rope_scale<<<(S_LEN * NH * 64) / 256, 256, 0, stream>>>(Qb, NH, prior, 0);
  rope_scale<<<(S_LEN * NKV * 64) / 256, 256, 0, stream>>>(Kbuf, NKV, prior, 1);
  v_scale_transpose<<<(S_LEN * KVW) / 256, 256, 0, stream>>>(Vbuf, VTb, prior);

  // 4) fused flash attention (double-buffered TDM -> LDS -> WMMA)
  attn_fused<<<dim3(S_LEN / 16, NH), 32, 0, stream>>>(
      Qb, Kbuf, VTb, prior, hm1, hm2, Ob);

  // 5) output projection -> fp32 d_out
  gemm_bf16_xwt<<<dim3(HID / 64, S_LEN / 32), 32, 0, stream>>>(
      Ob, wo_bf, d_out, S_LEN, HID, HID, 1);
}
